// Generator_90520730731266
// MI455X (gfx1250) — matrix operand order repaired and checked by reading, not compiled, not run on hardware
//
#include <hip/hip_runtime.h>
#include <hip/hip_bf16.h>

typedef __attribute__((ext_vector_type(16))) _Float16 v16h;
typedef __attribute__((ext_vector_type(8)))  _Float16 v8h;
typedef __attribute__((ext_vector_type(8)))  float    v8f;
typedef __attribute__((ext_vector_type(4)))  int      v4i;

typedef __attribute__((address_space(1))) v4i g_v4i;   // global int4
typedef __attribute__((address_space(3))) v4i l_v4i;   // LDS int4

#define B_   256
#define T_   512
#define F_   64
#define U1_  256
#define U2_  128
#define OUT_ 24

// CDNA5 async global->LDS path (ASYNCcnt), with portable fallback.
#if defined(__has_builtin)
#  if __has_builtin(__builtin_amdgcn_global_load_async_to_lds_b128) && \
      __has_builtin(__builtin_amdgcn_s_wait_asynccnt)
#    define USE_ASYNC_LDS 1
#  endif
#endif
#ifndef USE_ASYNC_LDS
#  define USE_ASYNC_LDS 0
#endif

static __device__ __forceinline__ g_v4i* to_global_v4i(const void* p) {
  // Flat global address bits == AS(1) address bits.
  return (g_v4i*)(unsigned long long)(uintptr_t)p;
}
static __device__ __forceinline__ l_v4i* to_lds_v4i(void* p) {
  // ISA aperture rule: the LDS address is the low 32 bits of the flat address.
  return (l_v4i*)(unsigned int)(uintptr_t)p;
}

static __device__ __forceinline__ float sigmoidf_(float x) {
  return 1.0f / (1.0f + __expf(-x));
}

// ---------------------------------------------------------------------------
// Pack a row-major f32 weight matrix W[K,N] into WMMA-B fragment layout (f16).
// Fragment (kt,nt) covers K in [kt*32, kt*32+32), N in [nt*16, nt*16+16).
// Per 16x16x32 f16 B layout: lane l (0..15) holds column n16=l, K=0..7 & 16..23 of the
// fragment; lanes 16..31 hold K=8..15 & 24..31 of the same column.
// Stored flat: out[(nt*KT+kt)*512 + lane*16 + s].
// ---------------------------------------------------------------------------
__global__ __launch_bounds__(256) void pack_b_kernel(const float* __restrict__ W,
                                                     _Float16* __restrict__ out,
                                                     int K, int N) {
  int tid = blockIdx.x * 256 + threadIdx.x;
  if (tid >= K * N) return;
  int KT   = K >> 5;
  int fidx = tid >> 9;
  int r    = tid & 511;
  int lane = r >> 4;
  int s    = r & 15;
  int nt   = fidx / KT;
  int kt   = fidx - nt * KT;
  int n    = nt * 16 + (lane & 15);
  int k    = kt * 32 + (s >> 3) * 16 + (lane >> 4) * 8 + (s & 7);
  out[tid] = (_Float16)W[(size_t)k * N + n];
}

// ---------------------------------------------------------------------------
// GEMM: C[M,ldc] = A[M,lda](f32) * Bpacked + bias  (one wave per 16x16 tile).
// A fragment (16x32 f16): lane m (0..15) = row m, K 0..7 & 16..23;
// lanes 16..31 = row m, K 8..15 & 24..31.
// ---------------------------------------------------------------------------
__global__ __launch_bounds__(256) void gemm_a32_kernel(const float* __restrict__ A, int lda,
                                                       const _Float16* __restrict__ Bp,
                                                       const float* __restrict__ bias,
                                                       float* __restrict__ C, int ldc,
                                                       int NT, int KT) {
  int wave = (blockIdx.x * 256 + (int)threadIdx.x) >> 5;
  int lane = threadIdx.x & 31;
  int mt   = wave / NT;
  int nt   = wave - mt * NT;
  int m16  = lane & 15;
  int kh   = lane >> 4;
  const float* arow = A + (size_t)(mt * 16 + m16) * lda;
  v8f acc = {};
  for (int kt = 0; kt < KT; ++kt) {
    const float* ap = arow + kt * 32 + kh * 8;
    float4 f0 = *reinterpret_cast<const float4*>(ap);
    float4 f1 = *reinterpret_cast<const float4*>(ap + 4);
    float4 f2 = *reinterpret_cast<const float4*>(ap + 16);
    float4 f3 = *reinterpret_cast<const float4*>(ap + 20);
    float lo[8] = {f0.x, f0.y, f0.z, f0.w, f1.x, f1.y, f1.z, f1.w};
    float hi[8] = {f2.x, f2.y, f2.z, f2.w, f3.x, f3.y, f3.z, f3.w};
    v16h a;
#pragma unroll
    for (int i = 0; i < 8; ++i) { a[i] = (_Float16)lo[i]; a[i + 8] = (_Float16)hi[i]; }
    v16h b = *reinterpret_cast<const v16h*>(Bp + (size_t)(nt * KT + kt) * 512 + lane * 16);
    acc = __builtin_amdgcn_wmma_f32_16x16x32_f16(false, a, false, b, (short)0, acc, false, false);
  }
  float bv = bias[nt * 16 + m16];
#pragma unroll
  for (int r = 0; r < 8; ++r) {
    size_t row = (size_t)(mt * 16 + kh * 8 + r);
    C[row * ldc + nt * 16 + m16] = acc[r] + bv;
  }
}

// Same as above but A is f16 row-major.
__global__ __launch_bounds__(256) void gemm_a16_kernel(const _Float16* __restrict__ A, int lda,
                                                       const _Float16* __restrict__ Bp,
                                                       const float* __restrict__ bias,
                                                       float* __restrict__ C, int ldc,
                                                       int NT, int KT) {
  int wave = (blockIdx.x * 256 + (int)threadIdx.x) >> 5;
  int lane = threadIdx.x & 31;
  int mt   = wave / NT;
  int nt   = wave - mt * NT;
  int m16  = lane & 15;
  int kh   = lane >> 4;
  const _Float16* arow = A + (size_t)(mt * 16 + m16) * lda;
  v8f acc = {};
  for (int kt = 0; kt < KT; ++kt) {
    const _Float16* ap = arow + kt * 32 + kh * 8;
    v8h c0 = *reinterpret_cast<const v8h*>(ap);
    v8h c1 = *reinterpret_cast<const v8h*>(ap + 16);
    v16h a = __builtin_shufflevector(c0, c1, 0, 1, 2, 3, 4, 5, 6, 7,
                                     8, 9, 10, 11, 12, 13, 14, 15);
    v16h b = *reinterpret_cast<const v16h*>(Bp + (size_t)(nt * KT + kt) * 512 + lane * 16);
    acc = __builtin_amdgcn_wmma_f32_16x16x32_f16(false, a, false, b, (short)0, acc, false, false);
  }
  float bv = bias[nt * 16 + m16];
#pragma unroll
  for (int r = 0; r < 8; ++r) {
    size_t row = (size_t)(mt * 16 + kh * 8 + r);
    C[row * ldc + nt * 16 + m16] = acc[r] + bv;
  }
}

// ---------------------------------------------------------------------------
// GRU recurrence, persistent over T. Each workgroup owns 16 batch rows
// (recurrence independent per batch row => only workgroup-local barriers).
// Per step: (1) issue async global->LDS copies of this step's xg slice,
// (2) rg = h @ R via WMMA while the copies fly, (3) drain ASYNCcnt + barrier,
// (4) f32 VALU gating from LDS, update h.
// ---------------------------------------------------------------------------
template <int U>
__global__ __launch_bounds__(256) void gru_rec_kernel(const float* __restrict__ xg,
                                                      const float* __restrict__ rbias,
                                                      const _Float16* __restrict__ Rp,
                                                      _Float16* __restrict__ hseq,
                                                      float* __restrict__ hfinal,
                                                      int T) {
  constexpr int N3   = 3 * U;
  constexpr int KT   = U / 32;
  constexpr int NT   = N3 / 16;
  constexpr int NTPW = NT / 8;
  constexpr int NV   = (16 * N3) / 4;   // float4 transfers per step
  __shared__ __align__(16) _Float16 hsh[16 * U];     // current hidden state (f16)
  __shared__ __align__(16) float    rgsh[16 * N3];   // recurrent projection (f32)
  __shared__ __align__(16) float    xsh[16 * N3];    // staged input projection (f32)

  int tid  = threadIdx.x;
  int wave = tid >> 5;
  int lane = tid & 31;
  int m16  = lane & 15;
  int kh   = lane >> 4;
  int bbase = blockIdx.x * 16;

  for (int i = tid; i < 16 * U; i += 256) hsh[i] = (_Float16)0.0f;
  __syncthreads();

  for (int t = 0; t < T; ++t) {
    // ---- stage xg[:,t,:] for this workgroup's 16 rows into LDS ----
#pragma unroll
    for (int i = 0; i < NV / 256; ++i) {
      int v    = tid + i * 256;
      int rrow = v / (N3 / 4);
      int c    = (v - rrow * (N3 / 4)) * 4;
      const float* gp = xg + ((size_t)(bbase + rrow) * T + t) * N3 + c;
      float* lp = &xsh[rrow * N3 + c];
#if USE_ASYNC_LDS
      __builtin_amdgcn_global_load_async_to_lds_b128(to_global_v4i(gp), to_lds_v4i(lp), 0, 0);
#else
      *reinterpret_cast<float4*>(lp) = *reinterpret_cast<const float4*>(gp);
#endif
    }

    // ---- load A fragments of h (shared across all N tiles of this wave) ----
    v16h af[KT];
#pragma unroll
    for (int kt = 0; kt < KT; ++kt) {
      const _Float16* hp = &hsh[m16 * U + kt * 32 + kh * 8];
      v8h c0 = *reinterpret_cast<const v8h*>(hp);
      v8h c1 = *reinterpret_cast<const v8h*>(hp + 16);
      af[kt] = __builtin_shufflevector(c0, c1, 0, 1, 2, 3, 4, 5, 6, 7,
                                       8, 9, 10, 11, 12, 13, 14, 15);
    }
    // ---- rg = h @ R for this wave's N tiles (overlaps the async copies) ----
#pragma unroll
    for (int q = 0; q < NTPW; ++q) {
      int nt = wave * NTPW + q;
      v8f acc = {};
#pragma unroll
      for (int kt = 0; kt < KT; ++kt) {
        v16h b = *reinterpret_cast<const v16h*>(Rp + (size_t)(nt * KT + kt) * 512 + lane * 16);
        acc = __builtin_amdgcn_wmma_f32_16x16x32_f16(false, af[kt], false, b,
                                                     (short)0, acc, false, false);
      }
      int col = nt * 16 + m16;
#pragma unroll
      for (int r = 0; r < 8; ++r) rgsh[(kh * 8 + r) * N3 + col] = acc[r];
    }
#if USE_ASYNC_LDS
    __builtin_amdgcn_s_wait_asynccnt(0);
#endif
    __syncthreads();

    // ---- gating (f32 VALU): z/r/h gates + state update, all from LDS ----
#pragma unroll
    for (int i = 0; i < (16 * U) / 256; ++i) {
      int idx  = tid + i * 256;
      int rrow = idx / U;
      int j    = idx - rrow * U;
      const float* xr = &xsh[rrow * N3];
      float xz = xr[j], xrg = xr[U + j], xh = xr[2 * U + j];
      float rz = rgsh[rrow * N3 + j]         + rbias[j];
      float rr = rgsh[rrow * N3 + U + j]     + rbias[U + j];
      float rh = rgsh[rrow * N3 + 2 * U + j] + rbias[2 * U + j];
      float z  = sigmoidf_(xz + rz);
      float rg = sigmoidf_(xrg + rr);
      float hh = tanhf(xh + rg * rh);
      float hp = (float)hsh[rrow * U + j];
      float hn = z * hp + (1.0f - z) * hh;
      hsh[rrow * U + j] = (_Float16)hn;
      if (hseq) hseq[((size_t)(bbase + rrow) * T + t) * U + j] = (_Float16)hn;
      if (hfinal && t == T - 1) hfinal[(size_t)(bbase + rrow) * U + j] = hn;
    }
    __syncthreads();
  }
}

// ---------------------------------------------------------------------------
// Dense head: h2[256,128] -> Dense(64) -> Dense(32) -> Dense(24). One block
// (64 threads) per batch row; tiny, VALU-only.
// ---------------------------------------------------------------------------
__global__ __launch_bounds__(64) void head_kernel(const float* __restrict__ h2,
                                                  const float* __restrict__ w3, const float* __restrict__ b3,
                                                  const float* __restrict__ w4, const float* __restrict__ b4,
                                                  const float* __restrict__ w5, const float* __restrict__ b5,
                                                  float* __restrict__ out) {
  __shared__ float s64[64];
  __shared__ float s32[32];
  int b = blockIdx.x;
  int j = threadIdx.x;
  const float* h = h2 + (size_t)b * U2_;
  float acc = b3[j];
  for (int k = 0; k < U2_; ++k) acc += h[k] * w3[k * 64 + j];
  s64[j] = acc;
  __syncthreads();
  if (j < 32) {
    float a2 = b4[j];
    for (int k = 0; k < 64; ++k) a2 += s64[k] * w4[k * 32 + j];
    s32[j] = a2;
  }
  __syncthreads();
  if (j < OUT_) {
    float a3 = b5[j];
    for (int k = 0; k < 32; ++k) a3 += s32[k] * w5[k * OUT_ + j];
    out[(size_t)b * OUT_ + j] = a3;
  }
}

// ---------------------------------------------------------------------------
extern "C" void kernel_launch(void* const* d_in, const int* in_sizes, int n_in,
                              void* d_out, int out_size, void* d_ws, size_t ws_size,
                              hipStream_t stream) {
  (void)in_sizes; (void)n_in; (void)out_size; (void)ws_size;
  const float* x  = (const float*)d_in[0];   // [B,T,F]
  const float* k1 = (const float*)d_in[1];   // [64,768]
  const float* r1 = (const float*)d_in[2];   // [256,768]
  const float* b1 = (const float*)d_in[3];   // [2,768]
  const float* k2 = (const float*)d_in[4];   // [256,384]
  const float* r2 = (const float*)d_in[5];   // [128,384]
  const float* b2 = (const float*)d_in[6];   // [2,384]
  const float* w3 = (const float*)d_in[7];
  const float* b3 = (const float*)d_in[8];
  const float* w4 = (const float*)d_in[9];
  const float* b4 = (const float*)d_in[10];
  const float* w5 = (const float*)d_in[11];
  const float* b5 = (const float*)d_in[12];
  float* out = (float*)d_out;

  const size_t BT = (size_t)B_ * T_;                 // 131072
  char* ws = (char*)d_ws;
  size_t off = 0;
  float*     xg1   = (float*)(ws + off);     off += BT * (3 * U1_) * sizeof(float);     // 403 MB
  float*     xg2   = (float*)(ws + off);     off += BT * (3 * U2_) * sizeof(float);     // 201 MB
  _Float16*  h1seq = (_Float16*)(ws + off);  off += BT * U1_ * sizeof(_Float16);        //  67 MB
  float*     h2fin = (float*)(ws + off);     off += (size_t)B_ * U2_ * sizeof(float);
  _Float16*  k1p   = (_Float16*)(ws + off);  off += (size_t)F_  * 3 * U1_ * sizeof(_Float16);
  _Float16*  r1p   = (_Float16*)(ws + off);  off += (size_t)U1_ * 3 * U1_ * sizeof(_Float16);
  _Float16*  k2p   = (_Float16*)(ws + off);  off += (size_t)U1_ * 3 * U2_ * sizeof(_Float16);
  _Float16*  r2p   = (_Float16*)(ws + off);  off += (size_t)U2_ * 3 * U2_ * sizeof(_Float16);

  // 1) Pack weights into WMMA B-fragment layout (f16).
  pack_b_kernel<<<(F_  * 3 * U1_) / 256, 256, 0, stream>>>(k1, k1p, F_,  3 * U1_);
  pack_b_kernel<<<(U1_ * 3 * U1_) / 256, 256, 0, stream>>>(r1, r1p, U1_, 3 * U1_);
  pack_b_kernel<<<(U1_ * 3 * U2_) / 256, 256, 0, stream>>>(k2, k2p, U1_, 3 * U2_);
  pack_b_kernel<<<(U2_ * 3 * U2_) / 256, 256, 0, stream>>>(r2, r2p, U2_, 3 * U2_);

  // 2) xg1 = x @ k1 + b1[0]   ([BT,64] x [64,768])
  {
    int NT = (3 * U1_) / 16, KT = F_ / 32;                  // 48, 2
    int tiles = (int)(BT / 16) * NT;                        // 393216
    gemm_a32_kernel<<<tiles / 8, 256, 0, stream>>>(x, F_, k1p, b1, xg1, 3 * U1_, NT, KT);
  }

  // 3) GRU1 recurrence over T; emits h1seq (f16).
  gru_rec_kernel<U1_><<<B_ / 16, 256, 0, stream>>>(xg1, b1 + 3 * U1_, r1p, h1seq, nullptr, T_);

  // 4) xg2 = h1seq @ k2 + b2[0]   ([BT,256] x [256,384])
  {
    int NT = (3 * U2_) / 16, KT = U1_ / 32;                 // 24, 8
    int tiles = (int)(BT / 16) * NT;                        // 196608
    gemm_a16_kernel<<<tiles / 8, 256, 0, stream>>>(h1seq, U1_, k2p, b2, xg2, 3 * U2_, NT, KT);
  }

  // 5) GRU2 recurrence; only final state needed.
  gru_rec_kernel<U2_><<<B_ / 16, 256, 0, stream>>>(xg2, b2 + 3 * U2_, r2p, nullptr, h2fin, T_);

  // 6) Dense head.
  head_kernel<<<B_, 64, 0, stream>>>(h2fin, w3, b3, w4, b4, w5, b5, out);
}
